// Model8_9620726743224
// MI455X (gfx1250) — matrix operand order repaired and checked by reading, not compiled, not run on hardware
//
#include <hip/hip_runtime.h>
#include <hip/hip_bf16.h>
#include <math.h>

typedef _Float16 half_t;
typedef __attribute__((ext_vector_type(16))) _Float16 v16h;
typedef __attribute__((ext_vector_type(8)))  float    v8f;

// ---------------------------------------------------------------------------
// Generic f16 WMMA GEMM: out[r,c] = act( sum_k A[r,aofs+k] * Bcm[c,k] + bias[c] )
//   A: f16, row stride lda, zero-padded so cols [aofs, aofs+KP) are valid
//   Bcm: f16 column-major [ncPad, KP], zero-padded (prep_w)
//   rows must be a multiple of 16 (true for all call sites here)
// One wave per 16x16 output tile. Uses v_wmma_f32_16x16x32_f16.
// ---------------------------------------------------------------------------
template <int KP>
__global__ __launch_bounds__(256)
void gemm_wmma_f16(const half_t* __restrict__ A, int lda, int aofs,
                   const half_t* __restrict__ Bcm,
                   const float* __restrict__ bias,
                   float* __restrict__ outF, int ldf,
                   half_t* __restrict__ outH, int ldh, int hofs,
                   int rows, int ncols, int relu)
{
  int wave = (blockIdx.x * blockDim.x + threadIdx.x) >> 5;
  int colTiles = (ncols + 15) >> 4;
  int rowTile = wave / colTiles;
  int colTile = wave - rowTile * colTiles;
  if (rowTile * 16 >= rows) return;               // wave-uniform: EXEC stays all-1 for WMMA

  int lane = threadIdx.x & 31;
  int hi   = lane >> 4;                            // lane group (0: lanes 0-15, 1: 16-31)
  int mn   = lane & 15;

  const half_t* Arow = A + (size_t)(rowTile * 16 + mn) * lda + aofs;
  const half_t* Bcol = Bcm + (size_t)(colTile * 16 + mn) * KP + (hi << 4);

  v8f acc = {};
#pragma unroll
  for (int c = 0; c < (KP >> 5); ++c) {
    // A fragment: 16x32 f16, vgpr r holds K={2r, +1} (+16 if hi), matching B's contiguous order
    union { v16h v; unsigned int u[8]; } a;
#pragma unroll
    for (int r = 0; r < 8; ++r) {
      int kb = (c << 5) + (hi << 4) + (r << 1);
      a.u[r] = *(const unsigned int*)(Arow + kb);  // 2 consecutive halfs, 4B aligned
    }
    // B fragment: 32x16 f16, lane n holds K = 16*hi + h (contiguous -> one 32B load)
    v16h b = *(const v16h*)(Bcol + (c << 5));
    acc = __builtin_amdgcn_wmma_f32_16x16x32_f16(false, a.v, false, b,
                                                 (short)0, acc, false, false);
  }

  int col = colTile * 16 + mn;
  if (col < ncols) {
    float bv = bias ? bias[col] : 0.f;
#pragma unroll
    for (int r = 0; r < 8; ++r) {
      int row = rowTile * 16 + r + (hi << 3);      // C layout: vgpr r -> M = r + 8*hi
      float v = acc[r] + bv;
      if (relu) v = fmaxf(v, 0.f);
      if (outF) outF[(size_t)row * ldf + col] = v;
      if (outH) outH[(size_t)row * ldh + hofs + col] = (half_t)v;
    }
  }
}

// Convert weight [K, nc] f32 row-major -> column-major f16 [ncPad, Kp], zero padded.
__global__ void prep_w(const float* __restrict__ W, half_t* __restrict__ out,
                       int K, int nc, int Kp, int ncPad)
{
  int t = blockIdx.x * blockDim.x + threadIdx.x;
  if (t >= ncPad * Kp) return;
  int n = t / Kp, k = t - n * Kp;
  out[t] = (k < K && n < nc) ? (half_t)W[k * nc + n] : (half_t)0.f;
}

// F[N,80] = [xb(0..9) | xa(10..19) | x_(20..29) | x1(30..44) | zeros(45..79)]
__global__ void init_features(const float* __restrict__ x1, half_t* __restrict__ F,
                              half_t* __restrict__ xcH, int N)
{
  int n = blockIdx.x * blockDim.x + threadIdx.x;
  if (n >= N) return;
  half_t* f = F + (size_t)n * 80;
#pragma unroll
  for (int j = 0; j < 30; ++j) f[j] = (half_t)0.f;
#pragma unroll
  for (int j = 0; j < 15; ++j) f[30 + j] = (half_t)x1[n * 15 + j];
#pragma unroll
  for (int j = 45; j < 80; ++j) f[j] = (half_t)0.f;
  half_t* xc = xcH + (size_t)n * 16;
#pragma unroll
  for (int j = 0; j < 16; ++j) xc[j] = (half_t)0.f;
}

__global__ void node_init(float* __restrict__ emax, float* __restrict__ denom,
                          float* __restrict__ agg, int N)
{
  int n = blockIdx.x * blockDim.x + threadIdx.x;
  if (n >= N) return;
  emax[n] = -INFINITY;
  denom[n] = 0.f;
#pragma unroll
  for (int g = 0; g < 10; ++g) agg[(size_t)n * 10 + g] = 0.f;
}

// float atomic max via signed/unsigned int ordering trick (init must be -inf)
__device__ inline void atomicMaxFloat(float* addr, float v)
{
  if (v >= 0.f) atomicMax((int*)addr, __float_as_int(v));
  else          atomicMin((unsigned int*)addr, (unsigned int)__float_as_int(v));
}

// Edge pass 1: e = leaky_relu(xl[src]+xr[dst]) . att ; segment-max into emax[dst]
__global__ void edge_attention(const int* __restrict__ e0, const int* __restrict__ e1,
                               int E, int Etot,
                               const float* __restrict__ xl, const float* __restrict__ xr,
                               const float* __restrict__ att,
                               float* __restrict__ ebuf, float* __restrict__ emax)
{
  int i = blockIdx.x * blockDim.x + threadIdx.x;
  if (i >= Etot) return;
  int src = (i < E) ? e0[i] : (i - E);   // self-loops appended after real edges
  int dst = (i < E) ? e1[i] : (i - E);
  const float* L = xl + (size_t)src * 16;
  const float* R = xr + (size_t)dst * 16;
  float s = 0.f;
#pragma unroll
  for (int g = 0; g < 10; ++g) {
    float v = L[g] + R[g];
    v = (v > 0.f) ? v : 0.2f * v;
    s += v * att[g];
  }
  ebuf[i] = s;
  atomicMaxFloat(&emax[dst], s);
}

// Edge pass 2: w = exp(e - emax[dst]); denom[dst]+=w; agg[dst] += w * xl[src]
__global__ void edge_aggregate(const int* __restrict__ e0, const int* __restrict__ e1,
                               int E, int Etot,
                               const float* __restrict__ xl,
                               const float* __restrict__ ebuf,
                               const float* __restrict__ emax,
                               float* __restrict__ denom, float* __restrict__ agg)
{
  int i = blockIdx.x * blockDim.x + threadIdx.x;
  if (i >= Etot) return;
  int src = (i < E) ? e0[i] : (i - E);
  int dst = (i < E) ? e1[i] : (i - E);
  float w = expf(ebuf[i] - emax[dst]);
  atomicAdd(&denom[dst], w);
  const float* L = xl + (size_t)src * 16;
#pragma unroll
  for (int g = 0; g < 10; ++g)
    atomicAdd(&agg[(size_t)dst * 10 + g], w * L[g]);
}

// out = relu(agg/denom + bias), written as f16 into dst slot
__global__ void gat_finalize(const float* __restrict__ agg, const float* __restrict__ denom,
                             const float* __restrict__ bias,
                             half_t* __restrict__ dst, int ldh, int ofs, int N)
{
  int n = blockIdx.x * blockDim.x + threadIdx.x;
  if (n >= N) return;
  float inv = 1.f / denom[n];
  half_t* o = dst + (size_t)n * ldh + ofs;
#pragma unroll
  for (int g = 0; g < 10; ++g) {
    float v = agg[(size_t)n * 10 + g] * inv + bias[g];
    o[g] = (half_t)fmaxf(v, 0.f);
  }
}

// attack features [M*A, 64] f16 (48 used + 16 zero pad)
__global__ void build_attack(const int* __restrict__ asrc, const int* __restrict__ adst,
                             const float* __restrict__ armies,
                             const half_t* __restrict__ xcH, const float* __restrict__ x1,
                             half_t* __restrict__ out, int total)
{
  int t = blockIdx.x * blockDim.x + threadIdx.x;
  if (t >= total) return;
  int s = asrc[t], d = adst[t];
  float am = armies[t];
  half_t* o = out + (size_t)t * 64;
  const half_t* xs = xcH + (size_t)s * 16;
  const half_t* xd = xcH + (size_t)d * 16;
#pragma unroll
  for (int g = 0; g < 10; ++g) { o[g] = xs[g]; o[10 + g] = xd[g]; }
  const float* s1 = x1 + (size_t)s * 15;
  const float* d1 = x1 + (size_t)d * 15;
#pragma unroll
  for (int j = 0; j < 12; ++j) o[20 + j] = (half_t)s1[3 + j];
#pragma unroll
  for (int j = 0; j < 14; ++j) o[32 + j] = (half_t)d1[1 + j];
  o[46] = (half_t)am;
  o[47] = (half_t)(0.6f * am - 0.7f * (d1[3] + d1[4]));
#pragma unroll
  for (int j = 48; j < 64; ++j) o[j] = (half_t)0.f;
}

// deploy features [M*D, 32] f16 (23 used + 9 zero pad)
__global__ void build_deploy(const int* __restrict__ tgt, const float* __restrict__ armies,
                             const half_t* __restrict__ xcH, const float* __restrict__ x1,
                             half_t* __restrict__ out, int total)
{
  int t = blockIdx.x * blockDim.x + threadIdx.x;
  if (t >= total) return;
  int d = tgt[t];
  half_t* o = out + (size_t)t * 32;
  const half_t* xd = xcH + (size_t)d * 16;
#pragma unroll
  for (int g = 0; g < 10; ++g) o[g] = xd[g];
  const float* d1 = x1 + (size_t)d * 15;
#pragma unroll
  for (int j = 0; j < 12; ++j) o[10 + j] = (half_t)d1[3 + j];
  o[22] = (half_t)armies[t];
#pragma unroll
  for (int j = 23; j < 32; ++j) o[j] = (half_t)0.f;
}

// value features [N, 32] f16: [xc(10) | x1(15) | x2(4) | zeros(3)]
__global__ void build_vfeat(const half_t* __restrict__ xcH, const float* __restrict__ x1,
                            const float* __restrict__ x2, half_t* __restrict__ out, int N)
{
  int n = blockIdx.x * blockDim.x + threadIdx.x;
  if (n >= N) return;
  half_t* o = out + (size_t)n * 32;
  const half_t* xc = xcH + (size_t)n * 16;
#pragma unroll
  for (int g = 0; g < 10; ++g) o[g] = xc[g];
#pragma unroll
  for (int j = 0; j < 15; ++j) o[10 + j] = (half_t)x1[n * 15 + j];
#pragma unroll
  for (int j = 0; j < 4; ++j) o[25 + j] = (half_t)x2[j];
  o[29] = o[30] = o[31] = (half_t)0.f;
}

// One wave per move: softmax-attention pool over A attack + D deploy orders -> pm[m]
__global__ void pool_orders(const float* __restrict__ oAtt, const float* __restrict__ oDep,
                            const float* __restrict__ oaW, const float* __restrict__ oab,
                            const float* __restrict__ ovW, const float* __restrict__ ovb,
                            float* __restrict__ pm, int M, int A, int D)
{
  int wave = (blockIdx.x * blockDim.x + threadIdx.x) >> 5;
  if (wave >= M) return;
  int lane = threadIdx.x & 31;
  float ob = oab[0], vb = ovb[0];
  float sA = -INFINITY, tA = 0.f, sD = -INFINITY, tD = 0.f;
  if (lane < A) {
    const float* r = oAtt + ((size_t)wave * A + lane) * 20;
    sA = ob; tA = vb;
#pragma unroll
    for (int k = 0; k < 20; ++k) { sA += r[k] * oaW[k]; tA += r[k] * ovW[k]; }
  }
  if (lane < D) {
    const float* r = oDep + ((size_t)wave * D + lane) * 20;
    sD = ob; tD = vb;
#pragma unroll
    for (int k = 0; k < 20; ++k) { sD += r[k] * oaW[k]; tD += r[k] * ovW[k]; }
  }
  float mx = fmaxf(sA, sD);
  for (int o = 16; o; o >>= 1) mx = fmaxf(mx, __shfl_xor(mx, o));
  float eA = (lane < A) ? expf(sA - mx) : 0.f;
  float eD = (lane < D) ? expf(sD - mx) : 0.f;
  float den = eA + eD;
  float num = eA * tA + eD * tD;
  for (int o = 16; o; o >>= 1) { den += __shfl_xor(den, o); num += __shfl_xor(num, o); }
  if (lane == 0) pm[wave] = num / den;
}

// scal layout: [0]=amax, [1]=wsum, [2..11]=acc
__global__ void init_scal(float* scal)
{
  scal[0] = -INFINITY;
  for (int i = 1; i < 12; ++i) scal[i] = 0.f;
}

__global__ void value_pass1(const float* __restrict__ vbuf, const float* __restrict__ vaW,
                            const float* __restrict__ vab,
                            float* __restrict__ anode, float* __restrict__ scal, int N)
{
  int n = blockIdx.x * blockDim.x + threadIdx.x;
  if (n >= N) return;
  const float* v = vbuf + (size_t)n * 20;
  float a = vab[0];
#pragma unroll
  for (int k = 0; k < 20; ++k) a += v[k] * vaW[k];
  anode[n] = a;
  atomicMaxFloat(&scal[0], a);
}

__global__ void value_pass2(const float* __restrict__ vbuf, const float* __restrict__ anode,
                            const float* __restrict__ vvW, const float* __restrict__ vvb,
                            float* __restrict__ scal, int N)
{
  __shared__ float sacc[11];
  int tid = threadIdx.x;
  if (tid < 11) sacc[tid] = 0.f;
  __syncthreads();
  int n = blockIdx.x * blockDim.x + tid;
  if (n < N) {
    float w = expf(anode[n] - scal[0]);
    atomicAdd(&sacc[0], w);
    const float* v = vbuf + (size_t)n * 20;
#pragma unroll
    for (int g = 0; g < 10; ++g) {
      float u = vvb[g];
#pragma unroll
      for (int k = 0; k < 20; ++k) u += v[k] * vvW[k * 10 + g];
      atomicAdd(&sacc[1 + g], w * u);
    }
  }
  __syncthreads();
  if (tid < 11) atomicAdd(&scal[1 + tid], sacc[tid]);
}

__global__ void value_final(const float* __restrict__ scal, const float* __restrict__ vlW,
                            const float* __restrict__ vlb, float* __restrict__ out)
{
  float s = vlb[0];
  float inv = 1.f / scal[1];
#pragma unroll
  for (int g = 0; g < 10; ++g) {
    float vg = fmaxf(scal[2 + g] * inv, 0.f);
    s += vg * vlW[g];
  }
  out[0] = tanhf(s);
}

// one block: log_softmax over pm -> out[1..M]
__global__ void pm_finalize(const float* __restrict__ pm, float* __restrict__ out, int M)
{
  __shared__ float sm[32];
  __shared__ float bc;
  int tid = threadIdx.x, nw = blockDim.x >> 5;
  float mx = -INFINITY;
  for (int i = tid; i < M; i += blockDim.x) mx = fmaxf(mx, pm[i]);
  for (int o = 16; o; o >>= 1) mx = fmaxf(mx, __shfl_xor(mx, o));
  if ((tid & 31) == 0) sm[tid >> 5] = mx;
  __syncthreads();
  if (tid < 32) {
    float v = (tid < nw) ? sm[tid] : -INFINITY;
    for (int o = 16; o; o >>= 1) v = fmaxf(v, __shfl_xor(v, o));
    if (tid == 0) bc = v;
  }
  __syncthreads();
  mx = bc;
  float s = 0.f;
  for (int i = tid; i < M; i += blockDim.x) s += expf(pm[i] - mx);
  for (int o = 16; o; o >>= 1) s += __shfl_xor(s, o);
  __syncthreads();
  if ((tid & 31) == 0) sm[tid >> 5] = s;
  __syncthreads();
  if (tid < 32) {
    float v = (tid < nw) ? sm[tid] : 0.f;
    for (int o = 16; o; o >>= 1) v += __shfl_xor(v, o);
    if (tid == 0) bc = mx + logf(v);
  }
  __syncthreads();
  float lse = bc;
  for (int i = tid; i < M; i += blockDim.x) out[1 + i] = pm[i] - lse;
}

// ---------------------------------------------------------------------------
extern "C" void kernel_launch(void* const* d_in, const int* in_sizes, int n_in,
                              void* d_out, int out_size, void* d_ws, size_t ws_size,
                              hipStream_t stream)
{
  const float* x1   = (const float*)d_in[0];
  const float* x2   = (const float*)d_in[1];
  const int*   edg  = (const int*)d_in[2];
  const int*   aSrc = (const int*)d_in[3];
  const int*   aDst = (const int*)d_in[4];
  const float* aArm = (const float*)d_in[5];
  const int*   dTgt = (const int*)d_in[6];
  const float* dArm = (const float*)d_in[7];

  const int N = in_sizes[0] / 15;
  const int E = in_sizes[2] / 2;
  const int M = out_size - 1;
  const int A = in_sizes[3] / M;
  const int D = in_sizes[6] / M;
  const int Etot = E + N;
  const int* e0 = edg;
  const int* e1 = edg + E;

  auto a256 = [](size_t x) { return (x + 255) & ~(size_t)255; };
  size_t off = 0;
  auto carve = [&](size_t bytes) -> char* {
    off = a256(off);
    char* p = (char*)d_ws + off;
    off += bytes;
    return p;
  };

  // persistent buffers
  half_t* F    = (half_t*)carve((size_t)N * 80 * sizeof(half_t));
  float*  emax = (float*)carve((size_t)N * sizeof(float));
  float*  den  = (float*)carve((size_t)N * sizeof(float));
  float*  agg  = (float*)carve((size_t)N * 10 * sizeof(float));
  half_t* xcH  = (half_t*)carve((size_t)N * 16 * sizeof(half_t));
  float*  anode= (float*)carve((size_t)N * sizeof(float));
  float*  pm   = (float*)carve((size_t)M * sizeof(float));
  float*  scal = (float*)carve(64 * sizeof(float));
  // column-major f16 padded weights
  half_t* wInit = (half_t*)carve(32 * 16 * 2);
  half_t* wG1l  = (half_t*)carve(32 * 16 * 2);
  half_t* wG1r  = (half_t*)carve(32 * 16 * 2);
  half_t* wG2l  = (half_t*)carve(64 * 16 * 2);
  half_t* wG2r  = (half_t*)carve(64 * 16 * 2);
  half_t* wG3l  = (half_t*)carve(64 * 16 * 2);
  half_t* wG3r  = (half_t*)carve(64 * 16 * 2);
  half_t* wVt   = (half_t*)carve(32 * 32 * 2);
  half_t* wAt   = (half_t*)carve(64 * 32 * 2);
  half_t* wDt   = (half_t*)carve(32 * 32 * 2);

  // aliased region: GAT-phase (xl/xr/e) vs head-phase buffers (disjoint lifetimes)
  size_t o1 = 0;
  size_t xl_o = o1; o1 += a256((size_t)N * 16 * 4);
  size_t xr_o = o1; o1 += a256((size_t)N * 16 * 4);
  size_t eb_o = o1; o1 += a256((size_t)Etot * 4);
  size_t o2 = 0;
  size_t af_o = o2; o2 += a256((size_t)M * A * 64 * 2);
  size_t df_o = o2; o2 += a256((size_t)M * D * 32 * 2);
  size_t oa_o = o2; o2 += a256((size_t)M * A * 20 * 4);
  size_t od_o = o2; o2 += a256((size_t)M * D * 20 * 4);
  size_t vf_o = o2; o2 += a256((size_t)N * 32 * 2);
  size_t vb_o = o2; o2 += a256((size_t)N * 20 * 4);
  char* reg = carve(o1 > o2 ? o1 : o2);
  float*  xl   = (float*)(reg + xl_o);
  float*  xr   = (float*)(reg + xr_o);
  float*  ebuf = (float*)(reg + eb_o);
  half_t* attF = (half_t*)(reg + af_o);
  half_t* depF = (half_t*)(reg + df_o);
  float*  oAtt = (float*)(reg + oa_o);
  float*  oDep = (float*)(reg + od_o);
  half_t* vF   = (half_t*)(reg + vf_o);
  float*  vbuf = (float*)(reg + vb_o);

  auto gridFor = [](long long t, int b) { return (int)((t + b - 1) / b); };
  auto prep = [&](int widx, half_t* dst, int K, int nc, int Kp, int ncPad) {
    int t = ncPad * Kp;
    prep_w<<<gridFor(t, 256), 256, 0, stream>>>((const float*)d_in[widx], dst, K, nc, Kp, ncPad);
  };
  auto gemm = [&](const half_t* Aq, int lda, int aofs, const half_t* B, int Kp,
                  const float* bias, float* oF, int ldf, half_t* oH, int ldh, int hofs,
                  long long rows, int ncols, int relu) {
    int colT = (ncols + 15) / 16;
    long long waves = (rows / 16) * colT;
    int blocks = (int)((waves * 32 + 255) / 256);
    if (Kp == 32)
      gemm_wmma_f16<32><<<blocks, 256, 0, stream>>>(Aq, lda, aofs, B, bias, oF, ldf,
                                                    oH, ldh, hofs, (int)rows, ncols, relu);
    else
      gemm_wmma_f16<64><<<blocks, 256, 0, stream>>>(Aq, lda, aofs, B, bias, oF, ldf,
                                                    oH, ldh, hofs, (int)rows, ncols, relu);
  };

  // 1) weight conversion
  prep(8,  wInit, 15, 10, 32, 16);
  prep(10, wG1l,  25, 10, 32, 16);
  prep(11, wG1r,  25, 10, 32, 16);
  prep(14, wG2l,  35, 10, 64, 16);
  prep(15, wG2r,  35, 10, 64, 16);
  prep(18, wG3l,  45, 10, 64, 16);
  prep(19, wG3r,  45, 10, 64, 16);
  prep(22, wVt,   29, 20, 32, 32);
  prep(30, wAt,   48, 20, 64, 32);
  prep(32, wDt,   23, 20, 32, 32);

  // 2) feature buffer init (x1 into F cols 30..44, zeros elsewhere)
  init_features<<<gridFor(N, 256), 256, 0, stream>>>(x1, F, xcH, N);

  // 3) x_ = relu(x1 @ init_W + b) -> F cols 20..29 (reads cols 30..61, disjoint)
  gemm(F, 80, 30, wInit, 32, (const float*)d_in[9], nullptr, 0, F, 80, 20, N, 10, 1);

  // 4) three GATv2 layers
  struct Lyr { int aofs, Kp; half_t *Wl, *Wr; const float *att, *b; half_t* dst; int ldh, dofs; };
  Lyr L[3] = {
    {20, 32, wG1l, wG1r, (const float*)d_in[12], (const float*)d_in[13], F,   80, 10},
    {10, 64, wG2l, wG2r, (const float*)d_in[16], (const float*)d_in[17], F,   80, 0},
    { 0, 64, wG3l, wG3r, (const float*)d_in[20], (const float*)d_in[21], xcH, 16, 0},
  };
  for (int l = 0; l < 3; ++l) {
    gemm(F, 80, L[l].aofs, L[l].Wl, L[l].Kp, nullptr, xl, 16, nullptr, 0, 0, N, 10, 0);
    gemm(F, 80, L[l].aofs, L[l].Wr, L[l].Kp, nullptr, xr, 16, nullptr, 0, 0, N, 10, 0);
    node_init<<<gridFor(N, 256), 256, 0, stream>>>(emax, den, agg, N);
    edge_attention<<<gridFor(Etot, 256), 256, 0, stream>>>(e0, e1, E, Etot, xl, xr,
                                                           L[l].att, ebuf, emax);
    edge_aggregate<<<gridFor(Etot, 256), 256, 0, stream>>>(e0, e1, E, Etot, xl,
                                                           ebuf, emax, den, agg);
    gat_finalize<<<gridFor(N, 256), 256, 0, stream>>>(agg, den, L[l].b,
                                                      L[l].dst, L[l].ldh, L[l].dofs, N);
  }

  // 5) heads: build features, GEMMs
  build_attack<<<gridFor((long long)M * A, 256), 256, 0, stream>>>(aSrc, aDst, aArm, xcH, x1,
                                                                   attF, M * A);
  build_deploy<<<gridFor((long long)M * D, 256), 256, 0, stream>>>(dTgt, dArm, xcH, x1,
                                                                   depF, M * D);
  build_vfeat<<<gridFor(N, 256), 256, 0, stream>>>(xcH, x1, x2, vF, N);

  gemm(attF, 64, 0, wAt, 64, (const float*)d_in[31], oAtt, 20, nullptr, 0, 0,
       (long long)M * A, 20, 1);
  gemm(depF, 32, 0, wDt, 32, (const float*)d_in[33], oDep, 20, nullptr, 0, 0,
       (long long)M * D, 20, 1);
  gemm(vF, 32, 0, wVt, 32, (const float*)d_in[23], vbuf, 20, nullptr, 0, 0, N, 20, 1);

  // 6) order attention pool -> pm
  pool_orders<<<gridFor((long long)M * 32, 256), 256, 0, stream>>>(
      oAtt, oDep, (const float*)d_in[34], (const float*)d_in[35],
      (const float*)d_in[36], (const float*)d_in[37], pm, M, A, D);

  // 7) value head: softmax-attention pool over nodes
  init_scal<<<1, 1, 0, stream>>>(scal);
  value_pass1<<<gridFor(N, 256), 256, 0, stream>>>(vbuf, (const float*)d_in[24],
                                                   (const float*)d_in[25], anode, scal, N);
  value_pass2<<<gridFor(N, 256), 256, 0, stream>>>(vbuf, anode, (const float*)d_in[26],
                                                   (const float*)d_in[27], scal, N);
  value_final<<<1, 1, 0, stream>>>(scal, (const float*)d_in[28], (const float*)d_in[29],
                                   (float*)d_out);

  // 8) log_softmax(pm) -> out[1..M]
  pm_finalize<<<1, 1024, 0, stream>>>(pm, (float*)d_out, M);
}
